// Quantize_21174188769948
// MI455X (gfx1250) — matrix operand order repaired and checked by reading, not compiled, not run on hardware
//
#include <hip/hip_runtime.h>
#include <hip/hip_bf16.h>
#include <math.h>

#define B_DIM 32
#define D_DIM 256
#define T_DIM 2048
#define K_DIM 1024
#define N_ROWS (B_DIM * T_DIM)   // 65536

typedef __attribute__((ext_vector_type(16))) __bf16 v16bf;
typedef __attribute__((ext_vector_type(8)))  float  v8f;
typedef __attribute__((ext_vector_type(4)))  int    v4i;

union Frag {
    v16bf v;
    uint4 u[2];   // 32 bytes
};

// ---- CDNA5 async global->LDS path (ASYNCcnt), with safe fallback ----------
#if defined(__gfx1250__) && __has_builtin(__builtin_amdgcn_global_load_async_to_lds_b128)
#define USE_ASYNC_LDS 1
#else
#define USE_ASYNC_LDS 0
#endif

#if USE_ASYNC_LDS
// Builtin prototype (from compiler diagnostic): param0 = int4 addrspace(1)*
// (printed as "int __device__ *"), param1 = LDS pointer, then offset, cpol.
typedef __attribute__((address_space(1))) v4i gv4i;
typedef __attribute__((address_space(3))) v4i lv4i;
__device__ __forceinline__ void async_cp16(void* lds, const void* g) {
    __builtin_amdgcn_global_load_async_to_lds_b128(
        (gv4i*)g, (lv4i*)lds, /*offset=*/0, /*cpol=*/0);
}
template <int N> __device__ __forceinline__ void wait_async() {
#if __has_builtin(__builtin_amdgcn_s_wait_asynccnt)
    __builtin_amdgcn_s_wait_asynccnt(N);
#else
    asm volatile("s_wait_asynccnt %0" :: "i"(N) : "memory");
#endif
}
#endif

// Stage one 16 KiB tile-pair (32 codes x 256 d, bf16) into LDS.
// 256 threads x 4 x b128 = 16 KiB; per-wave ASYNCcnt += 4 on the async path.
__device__ __forceinline__ void stage16k(void* dst, const void* src, int tid) {
#pragma unroll
    for (int c = 0; c < 4; ++c) {
        const int off = tid * 16 + c * 4096;
#if USE_ASYNC_LDS
        async_cp16((char*)dst + off, (const char*)src + off);
#else
        *(uint4*)((char*)dst + off) = *(const uint4*)((const char*)src + off);
#endif
    }
}

// ---------------------------------------------------------------------------
// Kernel 1: embed [D,K] f32 -> embedT [K,D] bf16, e2[k] = sum_d e^2, zero hist
// ---------------------------------------------------------------------------
__global__ __launch_bounds__(256) void prep_embed_kernel(
    const float* __restrict__ embed, __hip_bfloat16* __restrict__ embedT,
    float* __restrict__ e2, float* __restrict__ counts)
{
    int k = blockIdx.x * 256 + threadIdx.x;
    if (k >= K_DIM) return;
    float acc = 0.f;
    for (int d = 0; d < D_DIM; ++d) {
        float v = embed[(size_t)d * K_DIM + k];   // coalesced across k
        acc += v * v;
        embedT[(size_t)k * D_DIM + d] = __float2bfloat16(v);
    }
    e2[k] = acc;
    counts[k] = 0.f;
}

// ---------------------------------------------------------------------------
// Kernel 2: x [B,D,T] f32 -> xbf [B*T, D] bf16 via 32x32 LDS tile transpose
// ---------------------------------------------------------------------------
__global__ __launch_bounds__(256) void prep_x_kernel(
    const float* __restrict__ x, __hip_bfloat16* __restrict__ xbf)
{
    __shared__ float tile[32][33];
    const int tilesT = T_DIM / 32;                 // 64
    const int tilesD = D_DIM / 32;                 // 8
    int bt = blockIdx.x % tilesT;
    int bd = (blockIdx.x / tilesT) % tilesD;
    int b  = blockIdx.x / (tilesT * tilesD);
    int t0 = bt * 32, d0 = bd * 32;
    int lt = threadIdx.x & 31;                     // 0..31
    int lr = threadIdx.x >> 5;                     // 0..7

#pragma unroll
    for (int i = 0; i < 4; ++i) {                  // coalesced in t
        int dr = lr + i * 8;
        tile[dr][lt] = x[((size_t)b * D_DIM + d0 + dr) * T_DIM + t0 + lt];
    }
    __syncthreads();
#pragma unroll
    for (int i = 0; i < 4; ++i) {                  // coalesced in d
        int tr = lr + i * 8;
        xbf[((size_t)b * T_DIM + t0 + tr) * D_DIM + d0 + lt] =
            __float2bfloat16(tile[lt][tr]);
    }
}

// ---------------------------------------------------------------------------
// Kernel 3: fused distance-GEMM (bf16 WMMA, LDS-staged B) + argmin + gather
// One wave = 16 rows; block = 8 waves = 128 rows. All 8 waves share the
// double-buffered 2x16KiB codebook tile in LDS (async-prefetched).
// ---------------------------------------------------------------------------
__global__ __launch_bounds__(256) void vq_wmma_kernel(
    const __hip_bfloat16* __restrict__ xbf,     // [N, D] bf16
    const __hip_bfloat16* __restrict__ embedT,  // [K, D] bf16
    const float* __restrict__ e2,               // [K]
    const float* __restrict__ embed,            // [D, K] f32 (original)
    float* __restrict__ outq,                   // [B, D, T]
    float* __restrict__ outIdx,                 // [N] (float-encoded)
    float* __restrict__ counts)                 // [K]
{
    __shared__ __hip_bfloat16 bufB[2][32 * D_DIM];  // 2 x 16 KiB
    __shared__ float lds_e2[K_DIM];                 // 4 KiB
    __shared__ int s_idx[8][16];

    const int tid     = threadIdx.x;
    const int lane    = tid & 31;
    const int w       = tid >> 5;
    const int rowBase = blockIdx.x * 128 + w * 16;
    const int m       = lane & 15;
    const int half    = lane >> 4;

    // preload |e|^2 into LDS (visible after the first in-loop barrier)
    for (int k = tid; k < K_DIM; k += 256) lds_e2[k] = e2[k];

    // --- A fragments: 16 rows x 256 d, 8 fragments of 16x32 bf16.
    // ISA layout: lane holds row m; elems 0..7 at k = 8*half + 0..7,
    // elems 8..15 at k = 16 + 8*half + 0..7 (per 32-wide k-chunk).
    Frag a[8];
    {
        const __hip_bfloat16* rowp = xbf + (size_t)(rowBase + m) * D_DIM;
        const int off2 = half * 8;
#pragma unroll
        for (int ks = 0; ks < 8; ++ks) {
            a[ks].u[0] = *(const uint4*)(rowp + ks * 32 + off2);
            a[ks].u[1] = *(const uint4*)(rowp + ks * 32 + 16 + off2);
        }
    }

    float bestv[8];
    int   besti[8];
#pragma unroll
    for (int v = 0; v < 8; ++v) { bestv[v] = 3.4e38f; besti[v] = 0; }

    // prologue: stage tile-pair 0 (codes 0..31)
    stage16k(bufB[0], embedT, tid);

    const int dofs = half * 8;   // B layout: lane holds col n; k = 8*half+0..7, 16+8*half+0..7
    for (int nt2 = 0; nt2 < K_DIM / 32; ++nt2) {     // 32 iterations
        const int cur = nt2 & 1;
        if (nt2 + 1 < K_DIM / 32)
            stage16k(bufB[cur ^ 1],
                     embedT + (size_t)(nt2 + 1) * 32 * D_DIM, tid);
#if USE_ASYNC_LDS
        if (nt2 + 1 < K_DIM / 32) wait_async<4>();   // current tile done,
        else                      wait_async<0>();   // prefetch still in flight
#endif
        __syncthreads();

        // two independent accumulator chains over the two 16-code subtiles
        v8f acc0 = {0.f, 0.f, 0.f, 0.f, 0.f, 0.f, 0.f, 0.f};
        v8f acc1 = {0.f, 0.f, 0.f, 0.f, 0.f, 0.f, 0.f, 0.f};
        const __hip_bfloat16* bp0 = &bufB[cur][(size_t)m * D_DIM];
        const __hip_bfloat16* bp1 = &bufB[cur][(size_t)(16 + m) * D_DIM];
#pragma unroll
        for (int ks = 0; ks < 8; ++ks) {
            Frag b0, b1;
            b0.u[0] = *(const uint4*)(bp0 + ks * 32 + dofs);
            b0.u[1] = *(const uint4*)(bp0 + ks * 32 + dofs + 16);
            b1.u[0] = *(const uint4*)(bp1 + ks * 32 + dofs);
            b1.u[1] = *(const uint4*)(bp1 + ks * 32 + dofs + 16);
            acc0 = __builtin_amdgcn_wmma_f32_16x16x32_bf16(
                false, a[ks].v, false, b0.v, (short)0, acc0, false, false);
            acc1 = __builtin_amdgcn_wmma_f32_16x16x32_bf16(
                false, a[ks].v, false, b1.v, (short)0, acc1, false, false);
        }

        const int   nc0 = nt2 * 32 + m;
        const float e20 = lds_e2[nc0];
        const float e21 = lds_e2[nc0 + 16];
#pragma unroll
        for (int v = 0; v < 8; ++v) {
            // distance (up to row-constant |x|^2): |e|^2 - 2 x.e
            float s0 = e20 - 2.0f * acc0[v];
            if (s0 < bestv[v]) { bestv[v] = s0; besti[v] = nc0; }
            float s1 = e21 - 2.0f * acc1[v];
            if (s1 < bestv[v]) { bestv[v] = s1; besti[v] = nc0 + 16; }
        }
        __syncthreads();   // all waves done reading before next overwrite
    }

    // --- reduce argmin across the 16 N-lanes of each half (rows v and v+8)
#pragma unroll
    for (int v = 0; v < 8; ++v) {
#pragma unroll
        for (int off = 8; off >= 1; off >>= 1) {
            float ov = __shfl_xor(bestv[v], off, 16);
            int   oi = __shfl_xor(besti[v], off, 16);
            if (ov < bestv[v] || (ov == bestv[v] && oi < besti[v])) {
                bestv[v] = ov; besti[v] = oi;   // jnp.argmin: first minimum
            }
        }
        if (m == 0) s_idx[w][v + half * 8] = besti[v];
    }
    __syncthreads();

    // --- indices (as float) + usage histogram
    if (lane < 16) {
        int idx = s_idx[w][lane];
        outIdx[rowBase + lane] = (float)idx;
        atomicAdd(&counts[idx], 1.0f);
    }

    // --- gather quantize: out[b, d, t] = embed[d, idx(row)], f32 precision.
    // Our 16 rows are 16 consecutive t within one b (16 | 2048).
    const int myIdx = s_idx[w][m];
    const int bI    = rowBase >> 11;            // rowBase / 2048
    const int tI    = (rowBase & 2047) + m;
    float* outBase = outq + (size_t)bI * D_DIM * T_DIM + tI;
#pragma unroll 4
    for (int d = half; d < D_DIM; d += 2) {     // 16 lanes coalesced in t
        outBase[(size_t)d * T_DIM] = embed[(size_t)d * K_DIM + myIdx];
    }
}

// ---------------------------------------------------------------------------
// Kernel 4: perplexity = exp(-sum p log(p + 1e-10)), p = counts / N
// ---------------------------------------------------------------------------
__global__ __launch_bounds__(256) void perplexity_kernel(
    const float* __restrict__ counts, float* __restrict__ outP)
{
    __shared__ float red[256];
    float s = 0.f;
    for (int k = threadIdx.x; k < K_DIM; k += 256) {
        float p = counts[k] * (1.0f / (float)N_ROWS);
        s -= p * logf(p + 1e-10f);
    }
    red[threadIdx.x] = s;
    __syncthreads();
    for (int st = 128; st > 0; st >>= 1) {
        if (threadIdx.x < st) red[threadIdx.x] += red[threadIdx.x + st];
        __syncthreads();
    }
    if (threadIdx.x == 0) outP[0] = expf(red[0]);
}

// ---------------------------------------------------------------------------
extern "C" void kernel_launch(void* const* d_in, const int* in_sizes, int n_in,
                              void* d_out, int out_size, void* d_ws, size_t ws_size,
                              hipStream_t stream) {
    (void)in_sizes; (void)n_in; (void)out_size; (void)ws_size;
    const float* x     = (const float*)d_in[0];   // [32, 256, 2048]
    const float* embed = (const float*)d_in[1];   // [256, 1024]
    float* out = (float*)d_out;

    // workspace layout (all 4KB-aligned): embedT bf16 | e2 | counts | xbf bf16
    char* ws = (char*)d_ws;
    __hip_bfloat16* embedT = (__hip_bfloat16*)ws;                       // 512 KiB
    float* e2     = (float*)(ws + (512u << 10));                        // 4 KiB
    float* counts = (float*)(ws + (512u << 10) + (4u << 10));           // 4 KiB
    __hip_bfloat16* xbf = (__hip_bfloat16*)(ws + (512u << 10) + (8u << 10)); // 32 MiB

    float* outq   = out;                                   // 16,777,216
    float* outIdx = out + (size_t)B_DIM * D_DIM * T_DIM;   // 65,536
    float* outP   = outIdx + (size_t)B_DIM * T_DIM;        // 1

    prep_embed_kernel<<<K_DIM / 256, 256, 0, stream>>>(embed, embedT, e2, counts);
    prep_x_kernel<<<B_DIM * (D_DIM / 32) * (T_DIM / 32), 256, 0, stream>>>(x, xbf);
    vq_wmma_kernel<<<N_ROWS / 128, 256, 0, stream>>>(xbf, embedT, e2, embed,
                                                     outq, outIdx, counts);
    perplexity_kernel<<<1, 256, 0, stream>>>(counts, outP);
}
